// ODEFunc_4827543241218
// MI455X (gfx1250) — matrix operand order repaired and checked by reading, not compiled, not run on hardware
//
#include <hip/hip_runtime.h>
#include <hip/hip_fp16.h>
#include <math.h>

#define NN 100000
#define EE 1600000
#define NTILES (NN / 16)   // 6250 (N divisible by 16)

typedef __attribute__((ext_vector_type(16))) _Float16 v16h;
typedef __attribute__((ext_vector_type(8)))  _Float16 v8h;
typedef __attribute__((ext_vector_type(8)))  float    v8f;

static __device__ __forceinline__ float softplusf_(float x) {
    return x > 20.0f ? x : log1pf(expf(x));
}

static __device__ __forceinline__ v16h cat16(v8h lo, v8h hi) {
    return __builtin_shufflevector(lo, hi, 0,1,2,3,4,5,6,7,8,9,10,11,12,13,14,15);
}

// ---------------------------------------------------------------------------
// Zero 32-bit words (degi + cursor are contiguous: 2*N ints)
// ---------------------------------------------------------------------------
__global__ void zero_b32(int* __restrict__ p, long n) {
    long i = (long)blockIdx.x * blockDim.x + threadIdx.x;
    if (i < n) p[i] = 0;
}

// ---------------------------------------------------------------------------
// CSR-by-destination build (turns the 205M-float-atomic scatter into a gather)
// ---------------------------------------------------------------------------
__global__ void count_deg(const int* __restrict__ edst, int* __restrict__ degi) {
    int e = blockIdx.x * blockDim.x + threadIdx.x;
    if (e < EE) atomicAdd(&degi[edst[e]], 1);
}

// Single-block exclusive scan of degi -> offs (1024 threads, ~98 elems each)
__global__ void scan_deg(const int* __restrict__ degi, int* __restrict__ offs) {
    __shared__ int partial[1024];
    const int tid   = threadIdx.x;
    const int chunk = (NN + 1023) / 1024;          // 98
    const int start = tid * chunk;
    const int end   = min(start + chunk, NN);
    int s = 0;
    for (int i = start; i < end; ++i) s += degi[i];
    partial[tid] = s;
    __syncthreads();
    // Hillis-Steele inclusive scan over the 1024 partials
    for (int off = 1; off < 1024; off <<= 1) {
        int v = (tid >= off) ? partial[tid - off] : 0;
        __syncthreads();
        partial[tid] += v;
        __syncthreads();
    }
    int base = (tid == 0) ? 0 : partial[tid - 1]; // exclusive base for this chunk
    for (int i = start; i < end; ++i) { offs[i] = base; base += degi[i]; }
}

__global__ void fill_csr(const int* __restrict__ esrc, const int* __restrict__ edst,
                         const int* __restrict__ offs, int* __restrict__ cursor,
                         int* __restrict__ elist) {
    int e = blockIdx.x * blockDim.x + threadIdx.x;
    if (e >= EE) return;
    int d   = edst[e];
    int pos = atomicAdd(&cursor[d], 1);
    elist[offs[d] + pos] = esrc[e];
}

// ---------------------------------------------------------------------------
// Gather: one wave32 per destination node; agg[d,:] = sum over neighbors h[s,:]
// h (51 MB) is L2-resident: ~820 MB of L2 loads replaces 205M global atomics.
// ---------------------------------------------------------------------------
__global__ void gather_agg(const float* __restrict__ u,
                           const int*   __restrict__ elist,
                           const int*   __restrict__ offs,
                           const int*   __restrict__ degi,
                           float* __restrict__ agg) {
    int gid  = blockIdx.x * blockDim.x + threadIdx.x;
    int d    = gid >> 5;
    int lane = gid & 31;
    if (d >= NN) return;
    const int base = offs[d];
    const int cnt  = degi[d];
    float4 acc = make_float4(0.f, 0.f, 0.f, 0.f);
    for (int j = 0; j < cnt; ++j) {
        const int s = elist[base + j];
        const float4 v = ((const float4*)(u + (size_t)s * 256 + 128))[lane];
        acc.x += v.x; acc.y += v.y; acc.z += v.z; acc.w += v.w;
    }
    ((float4*)(agg + (size_t)d * 128))[lane] = acc;
}

// ---------------------------------------------------------------------------
// Stage 1: h_ = softplus([h, agg/max(deg,1)] @ WA^T + bA)
// Writes u16 = [c, h_] as f16 for reuse by the four stage-2 GEMMs.
// ---------------------------------------------------------------------------
__launch_bounds__(256)
__global__ void stage1_kernel(const float* __restrict__ u,
                              const float* __restrict__ agg,
                              const int*   __restrict__ degi,
                              const float* __restrict__ WA,
                              const float* __restrict__ bA,
                              _Float16* __restrict__ u16) {
    __shared__ _Float16 wlds[128 * 256];  // WA as f16, row-major [n][k]
    const int tid = threadIdx.x;
    for (int i = tid; i < 128 * 256; i += 256) wlds[i] = (_Float16)WA[i];
    __syncthreads();

    const int lane = tid & 31;
    const int wave = tid >> 5;
    const int tile = blockIdx.x * 8 + wave;
    if (tile >= NTILES) return;

    const bool hih  = lane >= 16;
    const int  m    = tile * 16 + (lane & 15);   // A-fragment row
    const float inv = 1.0f / fmaxf((float)degi[m], 1.0f);
    const int  nrow = lane & 15;

    v8f acc[8] = {};

    for (int kt = 0; kt < 8; ++kt) {
        const int kbase = kt * 32;
        // A-fragment (16x32 f16): lanes 0-15 hold K={0..7,16..23}, lanes 16-31 K={8..15,24..31}
        v16h afrag;
        #pragma unroll
        for (int r2 = 0; r2 < 2; ++r2) {
            const int k = kbase + (r2 ? (hih ? 24 : 16) : (hih ? 8 : 0));
            float4 a, b;
            if (k < 128) {   // h half of x
                const float4* p = (const float4*)(u + (size_t)m * 256 + 128 + k);
                a = p[0]; b = p[1];
            } else {         // normalized agg half of x
                const float4* p = (const float4*)(agg + (size_t)m * 128 + (k - 128));
                a = p[0]; b = p[1];
                a.x *= inv; a.y *= inv; a.z *= inv; a.w *= inv;
                b.x *= inv; b.y *= inv; b.z *= inv; b.w *= inv;
            }
            afrag[r2*8+0] = (_Float16)a.x; afrag[r2*8+1] = (_Float16)a.y;
            afrag[r2*8+2] = (_Float16)a.z; afrag[r2*8+3] = (_Float16)a.w;
            afrag[r2*8+4] = (_Float16)b.x; afrag[r2*8+5] = (_Float16)b.y;
            afrag[r2*8+6] = (_Float16)b.z; afrag[r2*8+7] = (_Float16)b.w;
        }
        // B-fragment (32x16): half-wave column holds K={0..7,16..23} (lanes 0-15) / {8..15,24..31} (lanes 16-31)
        const int koff = kbase + (hih ? 8 : 0);
        #pragma unroll
        for (int nt = 0; nt < 8; ++nt) {
            const v8h* bp = (const v8h*)&wlds[(nt * 16 + nrow) * 256 + koff];
            v16h bfrag = cat16(bp[0], bp[2]);
            acc[nt] = __builtin_amdgcn_wmma_f32_16x16x32_f16(
                false, afrag, false, bfrag, (short)0, acc[nt], false, false);
        }
    }

    // epilogue: softplus + bias, store h_ (f16). C rows 0-7 in lanes 0-15, 8-15 in 16-31
    #pragma unroll
    for (int nt = 0; nt < 8; ++nt) {
        const int n = nt * 16 + nrow;
        const float b = bA[n];
        #pragma unroll
        for (int r = 0; r < 8; ++r) {
            const int mm = tile * 16 + r + (hih ? 8 : 0);
            u16[(size_t)mm * 256 + 128 + n] = (_Float16)softplusf_(acc[nt][r] + b);
        }
    }
    // copy c -> f16 half of u16
    for (int i = lane; i < 512; i += 32) {  // 16 rows * 32 float4
        const int mm = tile * 16 + (i >> 5);
        const int n4 = (i & 31) * 4;
        float4 cv = *(const float4*)(u + (size_t)mm * 256 + n4);
        _Float16* dst = u16 + (size_t)mm * 256 + n4;
        dst[0] = (_Float16)cv.x; dst[1] = (_Float16)cv.y;
        dst[2] = (_Float16)cv.z; dst[3] = (_Float16)cv.w;
    }
}

// ---------------------------------------------------------------------------
// Stage 2: four GEMM phases (Z -> Gc -> Fc -> Fh) over register-resident
// A-fragments of u_ = [c, h_]; elementwise combine; dc projection; write out.
// ---------------------------------------------------------------------------
__launch_bounds__(256)
__global__ void stage2_kernel(const float* __restrict__ u,
                              const _Float16* __restrict__ u16,
                              const float* __restrict__ WFc, const float* __restrict__ bFc,
                              const float* __restrict__ WFh, const float* __restrict__ bFh,
                              const float* __restrict__ WGc, const float* __restrict__ bGc,
                              const float* __restrict__ WZ,  const float* __restrict__ bZ,
                              float* __restrict__ out) {
    __shared__ _Float16 wlds[128 * 256];  // one f16 weight at a time (64 KB)
    const int tid   = threadIdx.x;
    const int lane  = tid & 31;
    const int wave  = tid >> 5;
    const int tile  = blockIdx.x * 8 + wave;
    const bool act  = tile < NTILES;
    const bool hih  = lane >= 16;
    const int  m    = tile * 16 + (lane & 15);
    const int  nrow = lane & 15;

    // Hoist the full A tile (16x256 f16 = 64 VGPRs/lane) once; reuse in 4 GEMMs.
    v16h afr[8];
    if (act) {
        #pragma unroll
        for (int kt = 0; kt < 8; ++kt) {
            const int klo = kt * 32 + (hih ? 8 : 0);
            const v8h* ap = (const v8h*)(u16 + (size_t)m * 256 + klo);
            afr[kt] = cat16(ap[0], ap[2]);  // K runs {klo..klo+7}, {klo+16..klo+23}
        }
    }

    auto load_w = [&](const float* __restrict__ W) {
        __syncthreads();
        for (int i = tid; i < 128 * 256; i += 256) wlds[i] = (_Float16)W[i];
        __syncthreads();
    };

    auto gemm = [&](v8f acc[8]) {
        for (int kt = 0; kt < 8; ++kt) {
            const int koff = kt * 32 + (hih ? 8 : 0);
            #pragma unroll
            for (int nt = 0; nt < 8; ++nt) {
                const v8h* bp = (const v8h*)&wlds[(nt * 16 + nrow) * 256 + koff];
                v16h bfrag = cat16(bp[0], bp[2]);
                acc[nt] = __builtin_amdgcn_wmma_f32_16x16x32_f16(
                    false, afr[kt], false, bfrag, (short)0, acc[nt], false, false);
            }
        }
    };

    v8f t[8] = {};

    // --- Z: t = tanh(u_ @ WZ^T + bZ) ---
    load_w(WZ);
    if (act) {
        v8f acc[8] = {};
        gemm(acc);
        #pragma unroll
        for (int nt = 0; nt < 8; ++nt) {
            const float b = bZ[nt * 16 + nrow];
            #pragma unroll
            for (int r = 0; r < 8; ++r) t[nt][r] = tanhf(acc[nt][r] + b);
        }
    }
    // --- Gc: t *= softplus(u_ @ WGc^T + bGc) ---
    load_w(WGc);
    if (act) {
        v8f acc[8] = {};
        gemm(acc);
        #pragma unroll
        for (int nt = 0; nt < 8; ++nt) {
            const float b = bGc[nt * 16 + nrow];
            #pragma unroll
            for (int r = 0; r < 8; ++r) t[nt][r] *= softplusf_(acc[nt][r] + b);
        }
    }
    // --- Fc: dc = t - softplus(.)*c, then project out c component ---
    load_w(WFc);
    if (act) {
        v8f acc[8] = {};
        gemm(acc);
        v8f cv[8], dc[8];
        #pragma unroll
        for (int nt = 0; nt < 8; ++nt) {
            const int n = nt * 16 + nrow;
            const float b = bFc[n];
            #pragma unroll
            for (int r = 0; r < 8; ++r) {
                const int mm = tile * 16 + r + (hih ? 8 : 0);
                const float c = u[(size_t)mm * 256 + n];
                cv[nt][r] = c;
                dc[nt][r] = t[nt][r] - softplusf_(acc[nt][r] + b) * c;
            }
        }
        // per-row reductions; a row lives in one 16-lane half, so xor masks < 16
        #pragma unroll
        for (int r = 0; r < 8; ++r) {
            float pdc = 0.f, pcc = 0.f;
            #pragma unroll
            for (int nt = 0; nt < 8; ++nt) {
                pdc += dc[nt][r] * cv[nt][r];
                pcc += cv[nt][r] * cv[nt][r];
            }
            #pragma unroll
            for (int off = 1; off < 16; off <<= 1) {
                pdc += __shfl_xor(pdc, off, 32);
                pcc += __shfl_xor(pcc, off, 32);
            }
            const float s = pdc / pcc;
            #pragma unroll
            for (int nt = 0; nt < 8; ++nt) dc[nt][r] -= s * cv[nt][r];
        }
        #pragma unroll
        for (int nt = 0; nt < 8; ++nt) {
            const int n = nt * 16 + nrow;
            #pragma unroll
            for (int r = 0; r < 8; ++r) {
                const int mm = tile * 16 + r + (hih ? 8 : 0);
                out[(size_t)mm * 256 + n] = dc[nt][r];
            }
        }
    }
    // --- Fh: dh = -softplus(u_ @ WFh^T + bFh) * h ---
    load_w(WFh);
    if (act) {
        v8f acc[8] = {};
        gemm(acc);
        #pragma unroll
        for (int nt = 0; nt < 8; ++nt) {
            const int n = nt * 16 + nrow;
            const float b = bFh[n];
            #pragma unroll
            for (int r = 0; r < 8; ++r) {
                const int mm = tile * 16 + r + (hih ? 8 : 0);
                const float h = u[(size_t)mm * 256 + 128 + n];
                out[(size_t)mm * 256 + 128 + n] = -softplusf_(acc[nt][r] + b) * h;
            }
        }
    }
}

// ---------------------------------------------------------------------------
extern "C" void kernel_launch(void* const* d_in, const int* in_sizes, int n_in,
                              void* d_out, int out_size, void* d_ws, size_t ws_size,
                              hipStream_t stream) {
    (void)in_sizes; (void)n_in; (void)out_size; (void)ws_size;
    // setup_inputs order: t, u, edge_src, edge_dst, WA, bA, WFc, bFc, WFh, bFh, WGc, bGc, WZ, bZ
    const float* u    = (const float*)d_in[1];
    const int*   esrc = (const int*)d_in[2];
    const int*   edst = (const int*)d_in[3];
    const float* WA   = (const float*)d_in[4];
    const float* bA   = (const float*)d_in[5];
    const float* WFc  = (const float*)d_in[6];
    const float* bFc  = (const float*)d_in[7];
    const float* WFh  = (const float*)d_in[8];
    const float* bFh  = (const float*)d_in[9];
    const float* WGc  = (const float*)d_in[10];
    const float* bGc  = (const float*)d_in[11];
    const float* WZ   = (const float*)d_in[12];
    const float* bZ   = (const float*)d_in[13];
    float* out = (float*)d_out;

    // workspace layout:
    // degi(N i32) | cursor(N i32) | offs(N i32) | elist(E i32) | agg(N*128 f32) | u16(N*256 f16)
    int* degi   = (int*)d_ws;
    int* cursor = degi + NN;
    int* offs   = cursor + NN;
    int* elist  = offs + NN;
    float*    agg = (float*)(elist + EE);              // (3N+E)*4 bytes: 16B-aligned
    _Float16* u16 = (_Float16*)(agg + (size_t)NN * 128);

    zero_b32<<<(2 * NN + 255) / 256, 256, 0, stream>>>(degi, 2L * NN); // degi + cursor

    count_deg<<<EE / 256, 256, 0, stream>>>(edst, degi);
    scan_deg<<<1, 1024, 0, stream>>>(degi, offs);
    fill_csr<<<EE / 256, 256, 0, stream>>>(esrc, edst, offs, cursor, elist);
    gather_agg<<<(NN * 32) / 256, 256, 0, stream>>>(u, elist, offs, degi, agg);

    const int blocks = (NTILES + 7) / 8;  // 8 node-tiles (128 nodes) per block
    stage1_kernel<<<blocks, 256, 0, stream>>>(u, agg, degi, WA, bA, u16);
    stage2_kernel<<<blocks, 256, 0, stream>>>(u, u16, WFc, bFc, WFh, bFh, WGc, bGc, WZ, bZ, out);
}
